// MotionCorrect_1494648619449
// MI455X (gfx1250) — matrix operand order repaired and checked by reading, not compiled, not run on hardware
//
#include <hip/hip_runtime.h>
#include <hip/hip_bf16.h>

// ---- problem constants (from reference) ----
#define HW      512
#define NB      64
#define CX      56          // (H - CD)/2
#define TDIM    380         // template crop size
#define TPAD_R  416         // padded template rows (so rlo+31 always in-bounds)
#define TPAD_C  384         // padded template cols (K padded to 12*32)
#define NS      21          // 2*MS+1 shifts per axis
#define NSH     (NS*NS)     // 441
#define NT      144400.0f   // 380*380
#define EPSV    1e-8f

typedef __attribute__((ext_vector_type(16))) __bf16 v16bf;
typedef __attribute__((ext_vector_type(8)))  float  v8f;

union ABf { uint4 q[2]; unsigned short u[16]; v16bf v; };

static __device__ __forceinline__ unsigned short f2bf(float f) {
    unsigned int u = __float_as_uint(f);
    u += 0x7FFFu + ((u >> 16) & 1u);     // round-to-nearest-even
    return (unsigned short)(u >> 16);
}

// ---------------- K1: template mean (single block) ----------------
__global__ void k_tstats(const float* __restrict__ tmpl, float* __restrict__ stats) {
    __shared__ float red[256];
    int tid = threadIdx.x;
    float s = 0.f;
    for (int i = tid; i < TDIM * TDIM; i += 256) {
        int r = i / TDIM, c = i % TDIM;
        s += tmpl[(66 + r) * HW + 66 + c];    // 66 = MS + CX
    }
    red[tid] = s; __syncthreads();
    for (int off = 128; off > 0; off >>= 1) {
        if (tid < off) red[tid] += red[tid + off];
        __syncthreads();
    }
    if (tid == 0) { stats[0] = red[0] / NT; stats[1] = 0.f; }
}

// ------- K2: zero-mean template -> padded bf16, accumulate t_var -------
__global__ void k_tfill(const float* __restrict__ tmpl, const float* __restrict__ stats,
                        unsigned short* __restrict__ Tbf, float* __restrict__ statsAcc) {
    __shared__ float red[256];
    int tid = threadIdx.x;
    int idx = blockIdx.x * 256 + tid;            // grid covers exactly 416*384
    int r = idx / TPAD_C, c = idx % TPAD_C;
    float mean = stats[0];
    float val = 0.f;
    if (r < TDIM && c < TDIM) val = tmpl[(66 + r) * HW + 66 + c] - mean;
    Tbf[idx] = f2bf(val);
    red[tid] = val * val; __syncthreads();
    for (int off = 128; off > 0; off >>= 1) {
        if (tid < off) red[tid] += red[tid + off];
        __syncthreads();
    }
    if (tid == 0) atomicAdd(&statsAcc[1], red[0]);
}

// ------- K3: WMMA correlation. 1 block = 1 batch, 8 waves, wave = image row -------
__global__ void __launch_bounds__(256) k_corr(const float* __restrict__ x,
                                              const unsigned short* __restrict__ Tbf,
                                              float* __restrict__ numOut) {
    __shared__ float sacc[NSH];
    __shared__ unsigned short rowbuf[8][416];
    int tid = threadIdx.x, lane = tid & 31, wv = tid >> 5;
    int b = blockIdx.x;

    for (int i = tid; i < NSH; i += 256) sacc[i] = 0.f;
    __syncthreads();

    const int m   = lane & 15;      // M row / N col within tile
    const int hi  = lane >> 4;      // half-wave select
    const int kbA = hi * 16;        // A: K base per half-wave
    const int kbB = hi * 16;        // B: K base per half-wave

    for (int yy = wv; yy < 400; yy += 8) {              // 50 rows per wave, uniform
        // stage image row (raw Xc; zero-mean not needed: t_zm sums to 0) as bf16
        const float* rowp = x + (((size_t)b * HW) + (CX + yy)) * HW + CX;
        for (int c = lane; c < 416; c += 32) {
            float v = (c < 400) ? rowp[c] : 0.f;
            rowbuf[wv][c] = f2bf(v);
        }
        int rlo = yy - 20; if (rlo < 0) rlo = 0;        // template rows [rlo, rlo+31]

        v8f acc00 = {}, acc01 = {}, acc10 = {}, acc11 = {};
        for (int kc = 0; kc < 12; ++kc) {
            int c0 = kc * 32;
            // A tiles: template rows, aligned 16B global loads (L2-resident)
            const unsigned short* a0p = Tbf + (size_t)(rlo + m)      * TPAD_C + c0 + kbA;
            const unsigned short* a1p = Tbf + (size_t)(rlo + 16 + m) * TPAD_C + c0 + kbA;
            __builtin_prefetch(a0p + TPAD_C, 0, 1);
            ABf A0, A1, B0, B1;
            A0.q[0] = *reinterpret_cast<const uint4*>(a0p);
            A0.q[1] = *reinterpret_cast<const uint4*>(a0p + 8);
            A1.q[0] = *reinterpret_cast<const uint4*>(a1p);
            A1.q[1] = *reinterpret_cast<const uint4*>(a1p + 8);
            // B tiles: Hankel slices of the staged row, B[k,n] = row[c0+k+dx0+n]
            int s0 = c0 + kbB + m;        // dx0 = 0
            int s1 = s0 + 16;             // dx0 = 16
#pragma unroll
            for (int i = 0; i < 16; ++i) {
                B0.u[i] = rowbuf[wv][s0 + i];
                B1.u[i] = rowbuf[wv][s1 + i];
            }
            acc00 = __builtin_amdgcn_wmma_f32_16x16x32_bf16(false, A0.v, false, B0.v, (short)0, acc00, false, false);
            acc01 = __builtin_amdgcn_wmma_f32_16x16x32_bf16(false, A0.v, false, B1.v, (short)0, acc01, false, false);
            acc10 = __builtin_amdgcn_wmma_f32_16x16x32_bf16(false, A1.v, false, B0.v, (short)0, acc10, false, false);
            acc11 = __builtin_amdgcn_wmma_f32_16x16x32_bf16(false, A1.v, false, B1.v, (short)0, acc11, false, false);
        }
        // C/D layout: VGPR j holds M = j + hi*8, N = lane&15
#pragma unroll
        for (int j = 0; j < 8; ++j) {
            int mrow = j + hi * 8;
            int r  = rlo + mrow;       int dy  = yy - r;
            int r2 = r + 16;           int dy2 = yy - r2;
            if (dy >= 0 && dy < NS && r < TDIM) {
                atomicAdd(&sacc[dy * NS + m], acc00[j]);
                if (m + 16 < NS) atomicAdd(&sacc[dy * NS + m + 16], acc01[j]);
            }
            if (dy2 >= 0 && dy2 < NS && r2 < TDIM) {
                atomicAdd(&sacc[dy2 * NS + m], acc10[j]);
                if (m + 16 < NS) atomicAdd(&sacc[dy2 * NS + m + 16], acc11[j]);
            }
        }
    }
    __syncthreads();
    for (int i = tid; i < NSH; i += 256) numOut[(size_t)b * NSH + i] = sacc[i];
}

// ------- K4: local window variance via sliding column sums. 1 block = 1 batch -------
__global__ void __launch_bounds__(256) k_var(const float* __restrict__ x,
                                             float* __restrict__ varOut) {
    __shared__ float cs1[400], cs2[400];
    __shared__ float w1[NS], w2[NS];
    int tid = threadIdx.x, b = blockIdx.x;
    const float* xb = x + (size_t)b * HW * HW;

    float s1[2] = {0.f, 0.f}, s2[2] = {0.f, 0.f};
    for (int j = 0; j < 2; ++j) {
        int c = tid + 256 * j;
        if (c < 400) {
            for (int r = 0; r < TDIM; ++r) {
                float v = xb[(size_t)(CX + r) * HW + CX + c];
                s1[j] += v; s2[j] += v * v;
            }
        }
    }
    for (int dy = 0; dy < NS; ++dy) {
        for (int j = 0; j < 2; ++j) {
            int c = tid + 256 * j;
            if (c < 400) { cs1[c] = s1[j]; cs2[c] = s2[j]; }
        }
        if (tid < NS) { w1[tid] = 0.f; w2[tid] = 0.f; }
        __syncthreads();
        if (tid < 252) {                       // 12 partial-summers per dx
            int dx = tid % NS, sub = tid / NS;
            int i0 = sub * 32, i1 = i0 + 32; if (i1 > TDIM) i1 = TDIM;
            float a = 0.f, q = 0.f;
            for (int i = i0; i < i1; ++i) { a += cs1[dx + i]; q += cs2[dx + i]; }
            atomicAdd(&w1[dx], a); atomicAdd(&w2[dx], q);
        }
        __syncthreads();
        if (tid < NS) {
            float S1 = w1[tid], S2 = w2[tid];
            float v = S2 - S1 * S1 * (1.0f / NT) + EPSV;
            if (v < 0.f) v = 0.f;
            varOut[(size_t)b * NSH + dy * NS + tid] = v;
        }
        __syncthreads();
        if (dy < NS - 1) {
            for (int j = 0; j < 2; ++j) {
                int c = tid + 256 * j;
                if (c < 400) {
                    float vn = xb[(size_t)(CX + TDIM + dy) * HW + CX + c];
                    float vo = xb[(size_t)(CX + dy) * HW + CX + c];
                    s1[j] += vn - vo; s2[j] += vn * vn - vo * vo;
                }
            }
        }
    }
}

// ------- K5: NCC, argmax, log-parabola subpixel. 1 wave per batch -------
__global__ void k_peak(const float* __restrict__ num, const float* __restrict__ var,
                       const float* __restrict__ stats, float* __restrict__ shifts) {
    __shared__ float ncc[NSH];
    __shared__ float bv[32]; __shared__ int bix[32];
    int lane = threadIdx.x, b = blockIdx.x;
    float tvar = stats[1] + EPSV;
    float best = -3.4e38f; int bidx = 0;
    for (int i = lane; i < NSH; i += 32) {
        float v = num[(size_t)b * NSH + i] * rsqrtf(tvar * var[(size_t)b * NSH + i]);
        if (!(v == v)) v = 0.f;
        ncc[i] = v;
        if (v > best) { best = v; bidx = i; }
    }
    bv[lane] = best; bix[lane] = bidx;
    __syncthreads();
    if (lane == 0) {
        float mb = bv[0]; int mi = bix[0];
        for (int l = 1; l < 32; ++l)
            if (bv[l] > mb || (bv[l] == mb && bix[l] < mi)) { mb = bv[l]; mi = bix[l]; }
        int shx = mi / NS, shy = mi % NS;
        auto G = [&](int dx, int dyy) -> float {
            int xi = (shx + dx + NS) % NS;        // emulate wrap indexing
            int yi = (shy + dyy + NS) % NS;
            return logf(ncc[xi * NS + yi]);
        };
        float lm = G(-1, 0), lp = G(1, 0), l4 = 4.f * G(0, 0);
        float um = G(0, -1), up = G(0, 1);
        float shxn = -(float)(shx - 10) - (lm - lp) / (2.f * lm - l4 + 2.f * lp);
        float shyn = -(float)(shy - 10) - (um - up) / (2.f * um - l4 + 2.f * up);
        shifts[2 * b] = shxn; shifts[2 * b + 1] = shyn;
    }
}

// ------- K6: bilinear warp + transpose, LDS-tiled so both sides coalesce -------
__global__ void __launch_bounds__(256) k_warp(const float* __restrict__ x,
                                              const float* __restrict__ shifts,
                                              float* __restrict__ out) {
    __shared__ float tile[33 * 34];
    int tx = threadIdx.x, ty = threadIdx.y;
    int h0 = blockIdx.x * 32, w0 = blockIdx.y * 32, b = blockIdx.z;
    float dy = shifts[2 * b], dx = shifts[2 * b + 1];
    float fK = floorf(-dy), fL = floorf(-dx);
    int K = (int)fK, L = (int)fL;
    float wr = (-dy) - fK, wc = (-dx) - fL;     // floor(h-dy) = h + floor(-dy), h integer
    const float* xb = x + (size_t)b * HW * HW;
    int tid = ty * 32 + tx;
    for (int idx = tid; idx < 33 * 33; idx += 256) {
        int rl = idx / 33, cl = idx % 33;
        int gr = h0 + K + rl, gc = w0 + L + cl;
        float v = 0.f;
        if (gr >= 0 && gr < HW && gc >= 0 && gc < HW) v = xb[(size_t)gr * HW + gc];
        tile[rl * 34 + cl] = v;
    }
    __syncthreads();
    float w00 = (1.f - wr) * (1.f - wc), w01 = (1.f - wr) * wc;
    float w10 = wr * (1.f - wc),         w11 = wr * wc;
    for (int wl = ty; wl < 32; wl += 8) {
        float v00 = tile[tx * 34 + wl];
        float v01 = tile[tx * 34 + wl + 1];
        float v10 = tile[(tx + 1) * 34 + wl];
        float v11 = tile[(tx + 1) * 34 + wl + 1];
        out[(size_t)b * HW * HW + (size_t)(w0 + wl) * HW + (h0 + tx)] =
            v00 * w00 + v01 * w01 + v10 * w10 + v11 * w11;
    }
}

extern "C" void kernel_launch(void* const* d_in, const int* in_sizes, int n_in,
                              void* d_out, int out_size, void* d_ws, size_t ws_size,
                              hipStream_t stream) {
    const float* x    = (const float*)d_in[0];   // (B, 512, 512, 1) f32
    const float* tmpl = (const float*)d_in[1];   // (512, 512) f32
    float* out = (float*)d_out;                  // (B, H*W) f32

    // workspace layout (all offsets 256B-aligned; ~546 KB total)
    char* w = (char*)d_ws;
    unsigned short* Tbf = (unsigned short*)w;                         // 416*384*2 = 319488
    float* stats  = (float*)(w + 319488);                             // [mean, t_var_sum]
    float* num    = (float*)(w + 319488 + 256);                       // 64*441 f32
    float* var    = (float*)(w + 319488 + 256 + 64 * NSH * 4);        // 64*441 f32
    float* shifts = (float*)(w + 319488 + 256 + 2 * 64 * NSH * 4);    // 64*2 f32

    k_tstats<<<1, 256, 0, stream>>>(tmpl, stats);
    k_tfill<<<(TPAD_R * TPAD_C) / 256, 256, 0, stream>>>(tmpl, stats, Tbf, stats);
    k_corr<<<NB, 256, 0, stream>>>(x, Tbf, num);
    k_var<<<NB, 256, 0, stream>>>(x, var);
    k_peak<<<NB, 32, 0, stream>>>(num, var, stats, shifts);
    k_warp<<<dim3(16, 16, NB), dim3(32, 8), 0, stream>>>(x, shifts, out);
}
